// SelectedLayersReadout_13804024889513
// MI455X (gfx1250) — matrix operand order repaired and checked by reading, not compiled, not run on hardware
//
#include <hip/hip_runtime.h>

// ---------------------------------------------------------------------------
// SelectedLayersReadout on MI455X (gfx1250):
//   logits[1024, 32000] = X[1024, 16384] * W^T + b
// X rows are 4 contiguous 4096-float slabs of all_hidden_states (layers
// 0,6,12,23), so the "gather" is a pure concat.
//
// Roofline: 2.3 GB mandatory traffic @ 23.3 TB/s => ~98us floor; 1.07 TFLOP.
// BF16 WMMA (f32 accumulate) makes compute negligible. W fp32 is converted
// inline through double-buffered LDS (one barrier per K-slab, staging of
// slab s+1 overlaps WMMA of slab s); X is pre-converted once into d_ws
// (32 MiB, L2-resident, reused 500x). A-fragments are rolling-double-buffered
// in registers so the global load for fragment i+1 issues before the 4 WMMAs
// consuming fragment i (prefetch crosses the barrier: A never touches LDS).
// ---------------------------------------------------------------------------

typedef __attribute__((ext_vector_type(16))) __bf16 v16bf;
typedef __attribute__((ext_vector_type(2)))  __bf16 bf16x2;
typedef __attribute__((ext_vector_type(8)))  float  v8f;

#define M_TOTAL 1024
#define N_TOTAL 32000
#define K_TOTAL 16384
#define BM 512              // block M tile (8 waves x 64 rows)
#define BN 64               // block N tile
#define BK 64               // K slab staged in LDS
#define THREADS 256         // 8 waves (wave32)
#define LDS_ROW_U32 36      // 64 bf16 (32 u32) + 4 u32 pad => 144B row stride
#define LDS_BUF_U32 (BN * LDS_ROW_U32)   // 2304 u32 per buffer

// f32 -> bf16 pair packed into one u32 (a low 16, b high 16). Native casts:
// RNE in hardware on gfx1250 (falls back to legalized sequence otherwise).
__device__ __forceinline__ unsigned int pack_bf16(float a, float b) {
  bf16x2 p;
  p.x = (__bf16)a;
  p.y = (__bf16)b;
  return __builtin_bit_cast(unsigned int, p);
}

union FragCast { struct { uint4 lo, hi; } u; v16bf v; };
__device__ __forceinline__ v16bf make_frag(uint4 a, uint4 b) {
  FragCast f; f.u.lo = a; f.u.hi = b; return f.v;
}

// A fragment (16-bit A 16x32 layout): per-lane octet at k+8*hi, then k+16+8*hi.
__device__ __forceinline__ v16bf load_a(const char* row_base, int koff) {
  const uint4* ap = (const uint4*)(row_base + (size_t)koff * 2);
  return make_frag(ap[0], ap[2]);
}

// ---------------------------------------------------------------------------
// Kernel 1: gather selected layers and convert X to bf16 in d_ws.
// X[m][l*4096 + j] = H[(SEL[l]*1024 + m)*4096 + j],  m = b*512 + s.
// ---------------------------------------------------------------------------
__global__ __launch_bounds__(256) void gather_cvt_x(
    const float* __restrict__ H, unsigned int* __restrict__ Xbf) {
  size_t tid = (size_t)blockIdx.x * 256 + threadIdx.x;   // 2,097,152 threads
  size_t p = tid * 8;                                    // flat bf16 index
  int m = (int)(p >> 14);                                // row (0..1023)
  int f = (int)(p & 16383);                              // feature in row
  int l = f >> 12;                                       // selected-layer slot
  int j = f & 4095;                                      // offset in slab
  int layer = (int)((0x170C0600u >> (l * 8)) & 0xFFu);   // {0,6,12,23}
  const float4* src = (const float4*)(H + ((size_t)(layer * 1024 + m) << 12) + j);
  float4 a = src[0], b = src[1];
  uint4 o;
  o.x = pack_bf16(a.x, a.y);
  o.y = pack_bf16(a.z, a.w);
  o.z = pack_bf16(b.x, b.y);
  o.w = pack_bf16(b.z, b.w);
  *(uint4*)(Xbf + (p >> 1)) = o;
}

// ---------------------------------------------------------------------------
// Kernel 2: bf16 WMMA GEMM. Grid (M/BM=2, N/BN=500), 256 threads (8 waves).
// Wave w computes rows [m0, m0+64) x cols [n_blk, n_blk+64): 4x4 WMMA tiles.
// ---------------------------------------------------------------------------
__global__ __launch_bounds__(THREADS) void readout_gemm(
    const unsigned short* __restrict__ Xbf,
    const float* __restrict__ W,
    const float* __restrict__ bias,
    float* __restrict__ out) {
  __shared__ unsigned int Blds[2 * LDS_BUF_U32];   // 18432 B, double-buffered

  const int tid  = threadIdx.x;
  const int lane = tid & 31;
  const int wave = tid >> 5;        // 0..7
  const int lo   = lane & 15;       // M (A) / N (B,C) index within tile
  const int hi   = lane >> 4;       // K-half selector

  const int m_blk = blockIdx.x * BM;
  const int n_blk = blockIdx.y * BN;
  const int m0    = m_blk + wave * 64;

  // W staging: thread -> (row sr of 64, 16-float group sg of 4)
  const int sr = tid >> 2;
  const int sg = tid & 3;
  const float* wsrc = W + (size_t)(n_blk + sr) * K_TOTAL + sg * 16;
  unsigned int* sdst = Blds + sr * LDS_ROW_U32 + sg * 8;

  // A row base pointers (bf16 rows of 16384 = 32KB stride)
  const char* arow[4];
#pragma unroll
  for (int t = 0; t < 4; ++t)
    arow[t] = (const char*)Xbf + (size_t)(m0 + t * 16 + lo) * (K_TOTAL * 2) + hi * 16;

  // B frag base in LDS: column n = q*16+lo, K octets at 8*hi and 16+8*hi
  const unsigned int* brow[4];
#pragma unroll
  for (int q = 0; q < 4; ++q)
    brow[q] = Blds + (q * 16 + lo) * LDS_ROW_U32 + hi * 4;

  v8f c[4][4] = {};   // 128 accumulator VGPRs

  // --- pipeline prologue -------------------------------------------------
  // stage slab 0 into buffer 0
  {
    float4 a0 = ((const float4*)wsrc)[0];
    float4 a1 = ((const float4*)wsrc)[1];
    float4 a2 = ((const float4*)wsrc)[2];
    float4 a3 = ((const float4*)wsrc)[3];
    uint4 s0, s1;
    s0.x = pack_bf16(a0.x, a0.y); s0.y = pack_bf16(a0.z, a0.w);
    s0.z = pack_bf16(a1.x, a1.y); s0.w = pack_bf16(a1.z, a1.w);
    s1.x = pack_bf16(a2.x, a2.y); s1.y = pack_bf16(a2.z, a2.w);
    s1.z = pack_bf16(a3.x, a3.y); s1.w = pack_bf16(a3.z, a3.w);
    *(uint4*)(sdst)     = s0;
    *(uint4*)(sdst + 4) = s1;
  }
  // W slab 1 into registers
  float4 r0 = ((const float4*)(wsrc + BK))[0];
  float4 r1 = ((const float4*)(wsrc + BK))[1];
  float4 r2 = ((const float4*)(wsrc + BK))[2];
  float4 r3 = ((const float4*)(wsrc + BK))[3];
  // first A fragment
  v16bf Acur = load_a(arow[0], 0);
  __syncthreads();

  // --- main loop: one barrier per slab, staging overlaps compute ---------
  for (int ks = 0; ks < K_TOTAL; ks += BK) {
    const int cur = (ks >> 6) & 1;        // buffer holding slab ks
    const int nxt = cur ^ 1;
    const int ks_next = (ks + BK < K_TOTAL) ? ks + BK : 0;  // wrap: harmless

    // stage slab ks+BK into buf[nxt] while computing from buf[cur]
    if (ks + BK < K_TOTAL) {              // uniform branch
      uint4 s0, s1;
      s0.x = pack_bf16(r0.x, r0.y); s0.y = pack_bf16(r0.z, r0.w);
      s0.z = pack_bf16(r1.x, r1.y); s0.w = pack_bf16(r1.z, r1.w);
      s1.x = pack_bf16(r2.x, r2.y); s1.y = pack_bf16(r2.z, r2.w);
      s1.z = pack_bf16(r3.x, r3.y); s1.w = pack_bf16(r3.z, r3.w);
      *(uint4*)(sdst + nxt * LDS_BUF_U32)     = s0;
      *(uint4*)(sdst + nxt * LDS_BUF_U32 + 4) = s1;
      if (ks + 2 * BK < K_TOTAL) {        // prefetch W slab ks+2*BK
        const float4* nw = (const float4*)(wsrc + ks + 2 * BK);
        r0 = nw[0]; r1 = nw[1]; r2 = nw[2]; r3 = nw[3];
      }
    }

#pragma unroll
    for (int ki = 0; ki < BK; ki += 32) { // two K=32 WMMA steps per slab
      v16bf Bf[4];
#pragma unroll
      for (int q = 0; q < 4; ++q) {
        const uint4* bp = (const uint4*)(brow[q] + cur * LDS_BUF_U32 + (ki >> 1));
        Bf[q] = make_frag(bp[0], bp[2]);
      }
#pragma unroll
      for (int t = 0; t < 4; ++t) {
        // issue the NEXT A-fragment load before consuming the current one
        v16bf Anext;
        if (t < 3)            Anext = load_a(arow[t + 1], ks + ki);
        else if (ki == 0)     Anext = load_a(arow[0], ks + 32);
        else                  Anext = load_a(arow[0], ks_next); // next slab
#pragma unroll
        for (int q = 0; q < 4; ++q)
          c[t][q] = __builtin_amdgcn_wmma_f32_16x16x32_bf16(
              false, Acur, false, Bf[q], (short)0, c[t][q], false, false);
        Acur = Anext;
      }
    }
    __syncthreads();   // buf[nxt] stores + buf[cur] consumption complete
  }

  // --- epilogue: bias add + store. VGPR v -> row v + 8*hi, col = lo ------
  float bq[4];
#pragma unroll
  for (int q = 0; q < 4; ++q) bq[q] = bias[n_blk + q * 16 + lo];

#pragma unroll
  for (int t = 0; t < 4; ++t) {
#pragma unroll
    for (int v = 0; v < 8; ++v) {
      int m = m0 + t * 16 + v + 8 * hi;
      float* orow = out + (size_t)m * N_TOTAL + n_blk + lo;
#pragma unroll
      for (int q = 0; q < 4; ++q)
        orow[q * 16] = c[t][q][v] + bq[q];
    }
  }
}

// ---------------------------------------------------------------------------
extern "C" void kernel_launch(void* const* d_in, const int* in_sizes, int n_in,
                              void* d_out, int out_size, void* d_ws, size_t ws_size,
                              hipStream_t stream) {
  (void)in_sizes; (void)n_in; (void)out_size; (void)ws_size;
  const float* H    = (const float*)d_in[0];   // [24,2,512,512,8] fp32
  const float* W    = (const float*)d_in[1];   // [32000,16384] fp32
  const float* bvec = (const float*)d_in[2];   // [32000] fp32
  float* out = (float*)d_out;                  // [2,512,32000] fp32
  unsigned int* Xbf = (unsigned int*)d_ws;     // 32 MiB bf16 X[1024][16384]

  // 16,777,216 bf16 elements / 8 per thread / 256 per block = 8192 blocks
  gather_cvt_x<<<dim3(8192), dim3(256), 0, stream>>>(H, Xbf);

  dim3 grid(M_TOTAL / BM, N_TOTAL / BN);       // (2, 500)
  readout_gemm<<<grid, dim3(THREADS), 0, stream>>>(
      (const unsigned short*)Xbf, W, bvec, out);
}
